// JiTBlock_90769838833760
// MI455X (gfx1250) — matrix operand order repaired and checked by reading, not compiled, not run on hardware
//
#include <hip/hip_runtime.h>

#define DEV __device__ __forceinline__

typedef __attribute__((ext_vector_type(16))) __bf16 v16bf;
typedef __attribute__((ext_vector_type(8)))  float  v8f;

struct __align__(16) U128 { unsigned int v[4]; };

union Frag {
  v16bf bf;
  U128  q[2];
};

DEV unsigned short f2bf(float f) {
  unsigned int u = __float_as_uint(f);
  unsigned int r = u + 0x7FFFu + ((u >> 16) & 1u);
  return (unsigned short)(r >> 16);
}

DEV float siluf(float x) { return x * (1.f / (1.f + __expf(-x))); }

DEV v8f vzero() {
  v8f z;
#pragma unroll
  for (int i = 0; i < 8; ++i) z[i] = 0.f;
  return z;
}

// Model constants
constexpr int Dm    = 1024;
constexpr int Sseq  = 1024;
constexpr int Mrows = 16384;     // B*S
constexpr int MLPH  = 2730;

// GEMM tiling
constexpr int BN = 64, BK = 32;
constexpr int LDA = 40, LDB = 40;   // bf16 element strides (16B-aligned rows)

// ---------------------------------------------------------------------------
// ada = silu(c) @ w_ada + b_ada      [16, 6144]
// ---------------------------------------------------------------------------
__global__ __launch_bounds__(256, 1)
void ada_kernel(const float* __restrict__ c, const float* __restrict__ w_ada,
                const float* __restrict__ b_ada, float* __restrict__ ada) {
  __shared__ float sc_[1024];
  const int b = blockIdx.y, tid = threadIdx.x;
#pragma unroll
  for (int i = 0; i < 4; ++i) {
    int k = tid + i * 256;
    float v = c[b * 1024 + k];
    sc_[k] = siluf(v);
  }
  __syncthreads();
  int j = blockIdx.x * 256 + tid;
  float acc = b_ada[j];
  for (int k = 0; k < 1024; ++k) acc += sc_[k] * w_ada[(size_t)k * 6144 + j];
  ada[b * 6144 + j] = acc;
}

// ---------------------------------------------------------------------------
// h = modulate(rmsnorm(x, nw), ada[sh], ada[sc])  -> bf16
// ---------------------------------------------------------------------------
__global__ __launch_bounds__(256, 1)
void norm_mod_kernel(const float* __restrict__ xin, const float* __restrict__ nw,
                     const float* __restrict__ ada, int sh_off, int sc_off,
                     unsigned short* __restrict__ out) {
  __shared__ float red[256];
  const int row = blockIdx.x, tid = threadIdx.x;
  const int b = row >> 10;
  const float* xr = xin + (size_t)row * Dm;
  float v4[4], s = 0.f;
#pragma unroll
  for (int i = 0; i < 4; ++i) {
    float v = xr[tid + i * 256];
    v4[i] = v;
    s += v * v;
  }
  red[tid] = s;
  __syncthreads();
  for (int o = 128; o > 0; o >>= 1) {
    if (tid < o) red[tid] += red[tid + o];
    __syncthreads();
  }
  float rms = rsqrtf(red[0] * (1.f / 1024.f) + 1e-6f);
#pragma unroll
  for (int i = 0; i < 4; ++i) {
    int j = tid + i * 256;
    float h = v4[i] * rms * nw[j];
    h = h * (1.f + ada[b * 6144 + sc_off + j]) + ada[b * 6144 + sh_off + j];
    out[(size_t)row * Dm + j] = f2bf(h);
  }
}

// ---------------------------------------------------------------------------
// bf16 WMMA GEMM, software-pipelined (global->reg prefetch, LDS double buffer,
// one barrier per k-step). Each wave owns (MT*16) rows x 64 cols.
// EPI 0 = qkv (+bias, q/k rmsnorm+rope, scatter to q/k/v[B,H,S,HD])  (MT=1)
// EPI 1 = proj (+bias, xmid = x + g_a * .)                            (MT=2)
// EPI 2 = gate (dual-B: silu(x1+b1)*(x2+b2) -> bf16)                  (MT=1)
// EPI 3 = w3   (+bias, out = xmid + g_m * .)                          (MT=2)
// ---------------------------------------------------------------------------
template<int EPI, int MT>
__global__ __launch_bounds__(256, 1)
void gemm_bf16(const unsigned short* __restrict__ A, const float* __restrict__ Bg,
               const float* __restrict__ bias, int N, int K, int ldb, int dual_off,
               const float* __restrict__ resid, const float* __restrict__ ada, int ada_off,
               const float* __restrict__ qn_w, const float* __restrict__ kn_w,
               float* __restrict__ outF, unsigned short* __restrict__ outB,
               unsigned short* __restrict__ qws, unsigned short* __restrict__ kws,
               unsigned short* __restrict__ vws) {
  constexpr int NS  = (EPI == 2) ? 2 : 1;
  constexpr int BMk = MT * 128;
  extern __shared__ char smem[];
  unsigned short* sA = (unsigned short*)smem;        // 2 * BMk*LDA
  unsigned short* sB = sA + 2 * BMk * LDA;           // 2 * NS*BN*LDB
  float* sC = (float*)(sB + 2 * NS * BN * LDB);      // BMk*BN (EPI==0 only)

  const int tid  = threadIdx.x;
  const int wave = tid >> 5, lane = tid & 31;
  const int nrel = lane & 15, half = lane >> 4;
  const int mBase = blockIdx.y * BMk;
  const int nn    = blockIdx.x * BN;

  v8f acc[NS][MT][4];
#pragma unroll
  for (int s_ = 0; s_ < NS; ++s_)
#pragma unroll
    for (int mt = 0; mt < MT; ++mt)
#pragma unroll
      for (int t = 0; t < 4; ++t) acc[s_][mt][t] = vzero();

  unsigned int aReg[MT * 8];
  float bReg[NS][8];

  // global tile -> registers
  auto gstage = [&](int kk) {
#pragma unroll
    for (int i = 0; i < MT * 8; ++i) {
      int e = tid + i * 256;                // pair index over BMk x 16 pairs
      int r = e >> 4, kp = (e & 15) << 1;
      unsigned int val = 0u;
      if (kk + kp < K)
        val = *(const unsigned int*)(A + (size_t)(mBase + r) * K + kk + kp);
      aReg[i] = val;
    }
    for (int s_ = 0; s_ < NS; ++s_) {
#pragma unroll
      for (int i = 0; i < 8; ++i) {
        int e = tid + i * 256;
        int kb = e >> 6, n = e & 63;
        float v = 0.f;
        if ((kk + kb) < K && (nn + n) < N)
          v = Bg[(size_t)(kk + kb) * ldb + nn + n + (s_ ? dual_off : 0)];
        bReg[s_][i] = v;
      }
    }
  };

  // registers -> LDS buffer
  auto lstage = [&](int buf) {
    unsigned short* sAb = sA + buf * BMk * LDA;
    unsigned short* sBb = sB + buf * NS * BN * LDB;
#pragma unroll
    for (int i = 0; i < MT * 8; ++i) {
      int e = tid + i * 256;
      int r = e >> 4, kp = (e & 15) << 1;
      *(unsigned int*)(sAb + r * LDA + kp) = aReg[i];
    }
    for (int s_ = 0; s_ < NS; ++s_) {
#pragma unroll
      for (int i = 0; i < 8; ++i) {
        int e = tid + i * 256;
        int kb = e >> 6, n = e & 63;
        sBb[s_ * BN * LDB + n * LDB + kb] = f2bf(bReg[s_][i]);
      }
    }
  };

  // WMMA over one staged k-step
  auto compute = [&](int buf) {
    const unsigned short* sAb = sA + buf * BMk * LDA;
    const unsigned short* sBb = sB + buf * NS * BN * LDB;
    Frag af[MT];
#pragma unroll
    for (int mt = 0; mt < MT; ++mt) {
      const unsigned short* p = sAb + (wave * MT * 16 + mt * 16 + nrel) * LDA;
      af[mt].q[0] = *(const U128*)(p + half * 8);
      af[mt].q[1] = *(const U128*)(p + 16 + half * 8);
    }
#pragma unroll
    for (int s_ = 0; s_ < NS; ++s_)
#pragma unroll
      for (int t = 0; t < 4; ++t) {
        Frag bf_;
        const unsigned short* pb = sBb + s_ * BN * LDB + (t * 16 + nrel) * LDB;
        bf_.q[0] = *(const U128*)(pb + half * 8);
        bf_.q[1] = *(const U128*)(pb + 16 + half * 8);
#pragma unroll
        for (int mt = 0; mt < MT; ++mt)
          acc[s_][mt][t] = __builtin_amdgcn_wmma_f32_16x16x32_bf16(
              false, af[mt].bf, false, bf_.bf, (short)0, acc[s_][mt][t], false, false);
      }
  };

  const int nk = (K + BK - 1) / BK;
  gstage(0);
  for (int ki = 0; ki < nk; ++ki) {
    lstage(ki & 1);
    __syncthreads();
    if (ki + 1 < nk) {
      if (ki + 2 < nk) {
        // L2 prefetch of the B tile two steps ahead (global_prefetch_b8)
        int kk2 = (ki + 2) * BK;
        __builtin_prefetch(Bg + (size_t)(kk2 + (tid >> 6)) * ldb + nn + (tid & 63), 0, 1);
      }
      gstage((ki + 1) * BK);
    }
    compute(ki & 1);
  }

  // ------------------------------- epilogues -------------------------------
  if constexpr (EPI == 0) {
    __syncthreads();
#pragma unroll
    for (int mt = 0; mt < MT; ++mt)
#pragma unroll
      for (int t = 0; t < 4; ++t)
#pragma unroll
        for (int rr = 0; rr < 8; ++rr)
          sC[(wave * MT * 16 + mt * 16 + rr + 8 * half) * BN + t * 16 + nrel] =
              acc[0][mt][t][rr];
    __syncthreads();
    if (tid < BMk) {
      const int mg = mBase + tid;
      const int b = mg >> 10, spos = mg & 1023;
      const int tsel = nn >> 10, head = (nn >> 6) & 15;
      float vals[64];
#pragma unroll
      for (int d = 0; d < 64; ++d) vals[d] = sC[tid * BN + d] + bias[nn + d];
      if (tsel < 2) {
        const float* w = (tsel == 0) ? qn_w : kn_w;
        float ss = 0.f;
#pragma unroll
        for (int d = 0; d < 64; ++d) ss += vals[d] * vals[d];
        float rn = rsqrtf(ss * (1.f / 64.f) + 1e-6f);
#pragma unroll
        for (int d = 0; d < 64; ++d) vals[d] *= rn * w[d];
#pragma unroll
        for (int i = 0; i < 32; ++i) {
          float inv = __expf(-(float)i * (9.210340371976184f / 32.f));
          float fr = (float)spos * inv;
          float sn, cs;
          __sincosf(fr, &sn, &cs);
          float x1 = vals[i], x2 = vals[i + 32];
          vals[i]      = x1 * cs - x2 * sn;
          vals[i + 32] = x1 * sn + x2 * cs;
        }
      }
      unsigned short* dst = (tsel == 0) ? qws : ((tsel == 1) ? kws : vws);
      size_t base = ((size_t)(b * 16 + head) * Sseq + spos) * 64;
#pragma unroll
      for (int d = 0; d < 64; ++d) dst[base + d] = f2bf(vals[d]);
    }
  } else if constexpr (EPI == 2) {
#pragma unroll
    for (int mt = 0; mt < MT; ++mt)
#pragma unroll
      for (int t = 0; t < 4; ++t)
#pragma unroll
        for (int rr = 0; rr < 8; ++rr) {
          int mg = mBase + wave * MT * 16 + mt * 16 + rr + 8 * half;
          int j  = nn + t * 16 + nrel;
          if (j < N) {
            float x1 = acc[0][mt][t][rr] + bias[j];
            float x2 = acc[1][mt][t][rr] + bias[j + dual_off];
            outB[(size_t)mg * N + j] = f2bf(siluf(x1) * x2);
          }
        }
  } else {
#pragma unroll
    for (int mt = 0; mt < MT; ++mt)
#pragma unroll
      for (int t = 0; t < 4; ++t)
#pragma unroll
        for (int rr = 0; rr < 8; ++rr) {
          int mg = mBase + wave * MT * 16 + mt * 16 + rr + 8 * half;
          int j  = nn + t * 16 + nrel;
          float v = acc[0][mt][t][rr] + bias[j];
          int bidx = mg >> 10;
          size_t o = (size_t)mg * Dm + j;
          outF[o] = resid[o] + ada[bidx * 6144 + ada_off + j] * v;
        }
  }
}

// ---------------------------------------------------------------------------
// Flash attention: per (b,h), 64-query block, online softmax, WMMA QK^T / PV
// ---------------------------------------------------------------------------
__global__ __launch_bounds__(128, 1)
void attn_kernel(const unsigned short* __restrict__ Q, const unsigned short* __restrict__ K,
                 const unsigned short* __restrict__ V, unsigned short* __restrict__ O) {
  __shared__ unsigned short sQ[64 * 72], sK[64 * 72], sVt[64 * 72], sP[64 * 72];
  __shared__ float sS[64 * 64];
  __shared__ float sAlpha[64], sLi[64];

  const int tid = threadIdx.x, wave = tid >> 5, lane = tid & 31;
  const int nrel = lane & 15, half = lane >> 4;
  const int qt = blockIdx.x, bh = blockIdx.y;
  const size_t headBase = (size_t)bh * Sseq * 64;

  // load Q tile (64x64 bf16)
#pragma unroll
  for (int i = 0; i < 16; ++i) {
    int e = tid + i * 128;
    int r = e >> 5, dp = (e & 31) << 1;
    *(unsigned int*)(sQ + r * 72 + dp) =
        *(const unsigned int*)(Q + headBase + (size_t)(qt * 64 + r) * 64 + dp);
  }

  float mi = -1e30f, li = 0.f;
  v8f accO[4];
#pragma unroll
  for (int t = 0; t < 4; ++t) accO[t] = vzero();

  for (int kt = 0; kt < 16; ++kt) {
    __syncthreads();
    // stage K (natural) and V (transposed)
#pragma unroll
    for (int i = 0; i < 16; ++i) {
      int e = tid + i * 128;
      int r = e >> 5, dp = (e & 31) << 1;
      *(unsigned int*)(sK + r * 72 + dp) =
          *(const unsigned int*)(K + headBase + (size_t)(kt * 64 + r) * 64 + dp);
    }
#pragma unroll
    for (int i = 0; i < 32; ++i) {
      int e = tid + i * 128;
      int r = e >> 6, d = e & 63;
      sVt[d * 72 + r] = V[headBase + (size_t)(kt * 64 + r) * 64 + d];
    }
    __syncthreads();

    // S = Q * K^T   (each wave: 16 q rows x 64 keys)
    v8f accS[4];
#pragma unroll
    for (int t = 0; t < 4; ++t) accS[t] = vzero();
#pragma unroll
    for (int ks = 0; ks < 2; ++ks) {
      Frag a;
      a.q[0] = *(const U128*)(sQ + (wave * 16 + nrel) * 72 + ks * 32 + half * 8);
      a.q[1] = *(const U128*)(sQ + (wave * 16 + nrel) * 72 + ks * 32 + 16 + half * 8);
#pragma unroll
      for (int t = 0; t < 4; ++t) {
        Frag b;
        b.q[0] = *(const U128*)(sK + (t * 16 + nrel) * 72 + ks * 32 + half * 8);
        b.q[1] = *(const U128*)(sK + (t * 16 + nrel) * 72 + ks * 32 + 16 + half * 8);
        accS[t] = __builtin_amdgcn_wmma_f32_16x16x32_bf16(
            false, a.bf, false, b.bf, (short)0, accS[t], false, false);
      }
    }
#pragma unroll
    for (int t = 0; t < 4; ++t)
#pragma unroll
      for (int rr = 0; rr < 8; ++rr)
        sS[(wave * 16 + rr + 8 * half) * 64 + t * 16 + nrel] = accS[t][rr];
    __syncthreads();

    // online softmax (one thread per query row)
    if (tid < 64) {
      float mnew = mi;
      for (int j = 0; j < 64; ++j) mnew = fmaxf(mnew, sS[tid * 64 + j] * 0.125f);
      float alpha = __expf(mi - mnew);
      float l = li * alpha;
      for (int j = 0; j < 64; ++j) {
        float p = __expf(sS[tid * 64 + j] * 0.125f - mnew);
        l += p;
        sP[tid * 72 + j] = f2bf(p);
      }
      mi = mnew;
      li = l;
      sAlpha[tid] = alpha;
    }
    __syncthreads();

    // O = alpha*O + P*V
#pragma unroll
    for (int t = 0; t < 4; ++t)
#pragma unroll
      for (int rr = 0; rr < 8; ++rr)
        accO[t][rr] *= sAlpha[wave * 16 + rr + 8 * half];
#pragma unroll
    for (int ks = 0; ks < 2; ++ks) {
      Frag a;
      a.q[0] = *(const U128*)(sP + (wave * 16 + nrel) * 72 + ks * 32 + half * 8);
      a.q[1] = *(const U128*)(sP + (wave * 16 + nrel) * 72 + ks * 32 + 16 + half * 8);
#pragma unroll
      for (int t = 0; t < 4; ++t) {
        Frag b;
        b.q[0] = *(const U128*)(sVt + (t * 16 + nrel) * 72 + ks * 32 + half * 8);
        b.q[1] = *(const U128*)(sVt + (t * 16 + nrel) * 72 + ks * 32 + 16 + half * 8);
        accO[t] = __builtin_amdgcn_wmma_f32_16x16x32_bf16(
            false, a.bf, false, b.bf, (short)0, accO[t], false, false);
      }
    }
  }

  if (tid < 64) sLi[tid] = li;
  __syncthreads();

  const int b = bh >> 4, h = bh & 15;
#pragma unroll
  for (int t = 0; t < 4; ++t)
#pragma unroll
    for (int rr = 0; rr < 8; ++rr) {
      int mrow = wave * 16 + rr + 8 * half;
      float ov = accO[t][rr] / sLi[mrow];
      int s = qt * 64 + mrow, d = t * 16 + nrel;
      O[((size_t)(b * Sseq + s)) * Dm + h * 64 + d] = f2bf(ov);
    }
}

// ---------------------------------------------------------------------------
extern "C" void kernel_launch(void* const* d_in, const int* in_sizes, int n_in,
                              void* d_out, int out_size, void* d_ws, size_t ws_size,
                              hipStream_t stream) {
  const float* x      = (const float*)d_in[0];
  const float* c      = (const float*)d_in[1];
  const float* w_qkv  = (const float*)d_in[2];
  const float* b_qkv  = (const float*)d_in[3];
  const float* w_proj = (const float*)d_in[4];
  const float* b_proj = (const float*)d_in[5];
  const float* w12    = (const float*)d_in[6];
  const float* b12    = (const float*)d_in[7];
  const float* w3     = (const float*)d_in[8];
  const float* b3     = (const float*)d_in[9];
  const float* w_ada  = (const float*)d_in[10];
  const float* b_ada  = (const float*)d_in[11];
  const float* n1w    = (const float*)d_in[12];
  const float* n2w    = (const float*)d_in[13];
  const float* qnw    = (const float*)d_in[14];
  const float* knw    = (const float*)d_in[15];

  char* ws = (char*)d_ws;
  const size_t SZ_H  = (size_t)Mrows * Dm * 2;      // 32 MiB (bf16 activation)
  const size_t OFF_H = 512 * 1024;
  const size_t OFF_Q = OFF_H + SZ_H;
  const size_t OFF_K = OFF_Q + SZ_H;
  const size_t OFF_V = OFF_K + SZ_H;
  const size_t OFF_X = OFF_V + SZ_H;

  float* ada           = (float*)(ws);
  unsigned short* hbuf = (unsigned short*)(ws + OFF_H);   // h -> o -> h2 (reused)
  unsigned short* qws  = (unsigned short*)(ws + OFF_Q);
  unsigned short* kws  = (unsigned short*)(ws + OFF_K);
  unsigned short* vws  = (unsigned short*)(ws + OFF_V);
  unsigned short* gws  = (unsigned short*)(ws + OFF_Q);   // reuses q/k/v after attn
  float* xmid          = (float*)(ws + OFF_X);
  float* out           = (float*)d_out;

  // dynamic LDS: 2 * (BMk*LDA + NS*BN*LDB) bf16 elems (+ C tile for qkv)
  const size_t smQKV  = 2 * (size_t)(128 * LDA + 1 * BN * LDB) * 2 + (size_t)128 * BN * 4; // 63488
  const size_t smMT2  = 2 * (size_t)(256 * LDA + 1 * BN * LDB) * 2;                        // 51200
  const size_t smGate = 2 * (size_t)(128 * LDA + 2 * BN * LDB) * 2;                        // 40960

  // 1) adaLN params
  ada_kernel<<<dim3(24, 16), 256, 0, stream>>>(c, w_ada, b_ada, ada);
  // 2) h = modulate(rmsnorm(x))
  norm_mod_kernel<<<dim3(Mrows), 256, 0, stream>>>(x, n1w, ada, 0, 1024, hbuf);
  // 3) qkv GEMM + fused q/k rmsnorm + rope
  gemm_bf16<0, 1><<<dim3(48, 128), 256, smQKV, stream>>>(
      hbuf, w_qkv, b_qkv, 3072, 1024, 3072, 0,
      nullptr, nullptr, 0, qnw, knw, nullptr, nullptr, qws, kws, vws);
  // 4) attention -> o (bf16, reuses hbuf)
  attn_kernel<<<dim3(16, 256), 128, 0, stream>>>(qws, kws, vws, hbuf);
  // 5) proj GEMM + residual with g_a -> xmid
  gemm_bf16<1, 2><<<dim3(16, 64), 256, smMT2, stream>>>(
      hbuf, w_proj, b_proj, 1024, 1024, 1024, 0,
      x, ada, 2048, nullptr, nullptr, xmid, nullptr, nullptr, nullptr, nullptr);
  // 6) h2 = modulate(rmsnorm(xmid))
  norm_mod_kernel<<<dim3(Mrows), 256, 0, stream>>>(xmid, n2w, ada, 3072, 4096, hbuf);
  // 7) fused SwiGLU: g = silu(h2@w1 + b1) * (h2@w2 + b2)
  gemm_bf16<2, 1><<<dim3(43, 128), 256, smGate, stream>>>(
      hbuf, w12, b12, MLPH, 1024, 2 * MLPH, MLPH,
      nullptr, nullptr, 0, nullptr, nullptr, nullptr, gws, nullptr, nullptr, nullptr);
  // 8) out = xmid + g_m * (g@w3 + b3)
  gemm_bf16<3, 2><<<dim3(16, 64), 256, smMT2, stream>>>(
      gws, w3, b3, 1024, MLPH, 1024, 0,
      xmid, ada, 5120, nullptr, nullptr, out, nullptr, nullptr, nullptr, nullptr);
}
